// FALR6_35794257445097
// MI455X (gfx1250) — compile-verified, matrix operand order checked by reading, not run on hardware
//
#include <hip/hip_runtime.h>

// ---------------- problem constants ----------------
static constexpr int BATCH = 4;
static constexpr int NN    = 384;
static constexpr int JT    = NN / 16;                        // 24 j-tiles
static constexpr int ICH   = 12;                             // i-chunks
static constexpr int IPC   = NN / ICH;                       // 32 senders/chunk
static constexpr size_t OUT_TRI = (size_t)BATCH * NN * 128;  // ret elems before tri

// ---------------- workspace layout (dword units) ----------------
static constexpr size_t PK_M1  = 0;        // 32 frags * 256 dw
static constexpr size_t PK_M2  = 8192;
static constexpr size_t PK_ME  = 16384;
static constexpr size_t PK_O3  = 24576;
static constexpr size_t PK_O12 = 32768;    // 64 frags
static constexpr size_t PK_S   = 49152;    // 8 frags
static constexpr size_t PK_TE  = 51200;    // 4 frags
static constexpr int    NFRAG  = 204;

static constexpr size_t WS_GM  = 52224;    // B*128
static constexpr size_t WS_GT  = 52736;    // B*8 (padded)
static constexpr size_t WS_PN  = 52768;    // B*N*128
static constexpr size_t WS_PH  = 249376;
static constexpr size_t WS_R0  = 445984;
static constexpr size_t WS_S1  = 642592;   // B*N*8
static constexpr size_t WS_S2  = 654880;
static constexpr size_t WS_MSP = 667168;   // ICH * B*N*128 partial maxes
static constexpr size_t MSP_STRIDE = (size_t)BATCH * NN * 128;  // per-chunk

// bf16 LDS tile: 16 rows x 128 elems, row stride 68 dwords (136 bf16)
static constexpr int LSTR = 68;

// ---------------- WMMA types ----------------
typedef __attribute__((ext_vector_type(16))) __bf16 v16bf;
typedef __attribute__((ext_vector_type(8)))  float  v8f;

union Frag {
  v16bf    v;
  unsigned u[8];
};

// exact RNE pack (used only in one-time weight packing)
__device__ __forceinline__ unsigned cvt2_bf16_rne(float a, float b) {
  unsigned ua = __float_as_uint(a), ub = __float_as_uint(b);
  ua = (ua + 0x7FFFu + ((ua >> 16) & 1u)) >> 16;
  ub = (ub + 0x7FFFu + ((ub >> 16) & 1u)) >> 16;
  return (ua & 0xFFFFu) | (ub << 16);
}

// fast pack: +0x8000 round-to-nearest, single v_perm_b32 to merge high halves
__device__ __forceinline__ unsigned pack2_bf16(float a, float b) {
  unsigned ua = __float_as_uint(a) + 0x8000u;
  unsigned ub = __float_as_uint(b) + 0x8000u;
  return __builtin_amdgcn_perm(ub, ua, 0x07060302u);  // {ub.hi16, ua.hi16}
}

__device__ __forceinline__ v8f wmma_bf16(const Frag& a, const Frag& b, v8f c) {
  return __builtin_amdgcn_wmma_f32_16x16x32_bf16(false, a.v, false, b.v,
                                                 (short)0, c, false, false);
}

// A-fragment (16x32 bf16) straight from a bf16 LDS tile: 2x ds_load_b128.
__device__ __forceinline__ void buildA16(Frag& f, const unsigned* lds, int row,
                                         int ks, int hi) {
  const unsigned* p = lds + row * LSTR + ks * 16 + (hi ? 4 : 0);
#pragma unroll
  for (int v = 0; v < 4; ++v) f.u[v] = p[v];
#pragma unroll
  for (int v = 0; v < 4; ++v) f.u[4 + v] = p[8 + v];
}

__device__ __forceinline__ void loadPK(Frag& f, const unsigned* p) {
#pragma unroll
  for (int v = 0; v < 8; ++v) f.u[v] = p[v];
}

// Convert one thread's 16 consecutive f32 (4 float4) into 8 packed bf16 dwords
// and store to LDS tile (row = tid/8, col base = (tid%8)*16 elements).
__device__ __forceinline__ void stageTile16(unsigned* lds, int tid,
                                            const float4 r[4]) {
  unsigned* dst = lds + (tid >> 3) * LSTR + (tid & 7) * 8;
  uint4 lo, hi4;
  lo.x  = pack2_bf16(r[0].x, r[0].y); lo.y  = pack2_bf16(r[0].z, r[0].w);
  lo.z  = pack2_bf16(r[1].x, r[1].y); lo.w  = pack2_bf16(r[1].z, r[1].w);
  hi4.x = pack2_bf16(r[2].x, r[2].y); hi4.y = pack2_bf16(r[2].z, r[2].w);
  hi4.z = pack2_bf16(r[3].x, r[3].y); hi4.w = pack2_bf16(r[3].z, r[3].w);
  *(uint4*)dst       = lo;
  *(uint4*)(dst + 4) = hi4;
}

// =====================================================================
// Kernel A: pack weights to bf16 WMMA B-fragment layout; compute gM/gT.
// =====================================================================
__global__ void __launch_bounds__(256) kprep(
    const float* Wm1, const float* Wm2, const float* Wme,
    const float* Wo1, const float* Wo2, const float* Wo3,
    const float* Wtn1, const float* Wtn2, const float* Wth1, const float* Wth2,
    const float* Wte1, const float* Wte2,
    const float* Wmg, const float* Wtg1, const float* Wtg2,
    const float* graph,
    const float* bm1, const float* bm2, const float* bme, const float* bmg,
    const float* btn1, const float* btn2, const float* bth1, const float* bth2,
    const float* bte1, const float* bte2, const float* btg1, const float* btg2,
    float* wsf) {
  unsigned* wsu = (unsigned*)wsf;
  int j = blockIdx.x * blockDim.x + threadIdx.x;
  if (j < NFRAG * 32) {
    int fragIdx = j >> 5, lane = j & 31;
    int nlo = lane & 15, hi = lane >> 4;
    int mat, local, nct;
    size_t dwBase;
    if (fragIdx < 32)       { mat = 0; local = fragIdx;       nct = 8; dwBase = PK_M1; }
    else if (fragIdx < 64)  { mat = 1; local = fragIdx - 32;  nct = 8; dwBase = PK_M2; }
    else if (fragIdx < 96)  { mat = 2; local = fragIdx - 64;  nct = 8; dwBase = PK_ME; }
    else if (fragIdx < 128) { mat = 3; local = fragIdx - 96;  nct = 8; dwBase = PK_O3; }
    else if (fragIdx < 192) { mat = 4; local = fragIdx - 128; nct = 8; dwBase = PK_O12; }
    else if (fragIdx < 200) { mat = 5; local = fragIdx - 192; nct = 1; dwBase = PK_S; }
    else                    { mat = 6; local = fragIdx - 200; nct = 1; dwBase = PK_TE; }
    int ks = local / nct, ct = local % nct;
    int nc = ct * 16 + nlo;
    float vals[16];
#pragma unroll
    for (int e = 0; e < 16; ++e) {
      int k = ks * 32 + ((e < 8 ? e : e + 8) + (hi ? 8 : 0));
      float v;
      switch (mat) {
        case 0: v = Wm1[k * 128 + nc]; break;
        case 1: v = Wm2[k * 128 + nc]; break;
        case 2: v = Wme[k * 128 + nc]; break;
        case 3: v = Wo3[k * 128 + nc]; break;
        case 4: v = (k < 128) ? Wo1[k * 128 + nc] : Wo2[(k - 128) * 128 + nc]; break;
        case 5: {
          const float* W = (k < 128) ? ((nc < 8) ? Wtn1 : Wtn2)
                                     : ((nc < 8) ? Wth1 : Wth2);
          v = W[(k & 127) * 8 + (nc & 7)];
        } break;
        default: v = (nc < 8) ? (Wte1[k * 8 + nc] + Wte2[k * 8 + nc]) : 0.0f; break;
      }
      vals[e] = v;
    }
    unsigned* dst = wsu + dwBase + (size_t)local * 256 + (size_t)lane * 8;
#pragma unroll
    for (int vv = 0; vv < 8; ++vv)
      dst[vv] = cvt2_bf16_rne(vals[2 * vv], vals[2 * vv + 1]);
  } else if (j < NFRAG * 32 + 512) {
    int g = j - NFRAG * 32;
    int b = g >> 7, c = g & 127;
    float s = bmg[c] + bm1[c] + bm2[c] + bme[c];
    for (int k = 0; k < 128; ++k) s += graph[b * 128 + k] * Wmg[k * 128 + c];
    wsf[WS_GM + g] = s;
  } else if (j < NFRAG * 32 + 512 + 32) {
    int g = j - NFRAG * 32 - 512;
    int b = g >> 3, t = g & 7;
    float s = btg1[t] + btg2[t] + btn1[t] + btn2[t] + bth1[t] + bth2[t] + bte1[t] + bte2[t];
    for (int k = 0; k < 128; ++k)
      s += graph[b * 128 + k] * (Wtg1[k * 8 + t] + Wtg2[k * 8 + t]);
    wsf[WS_GT + g] = s;
  }
}

// =====================================================================
// Kernel B: node/hidden projections (Pn, Ph, R0, S1, S2) via WMMA.
// =====================================================================
__global__ void __launch_bounds__(128) knode(
    const float* node, const float* hidden,
    const float* bo1, const float* bo2, const float* bo3, float* wsf) {
  __shared__ unsigned ldsN[16 * LSTR];
  __shared__ unsigned ldsH[16 * LSTR];
  const unsigned* wsu = (const unsigned*)wsf;
  int b = blockIdx.x / JT, jt = blockIdx.x % JT, j0 = jt * 16;
  int tid = threadIdx.x;
  {
    const float4* srcN = (const float4*)(node + (size_t)(b * NN + j0) * 128);
    const float4* srcH = (const float4*)(hidden + (size_t)(b * NN + j0) * 128);
    float4 rN[4], rH[4];
#pragma unroll
    for (int q = 0; q < 4; ++q) { rN[q] = srcN[tid * 4 + q]; rH[q] = srcH[tid * 4 + q]; }
    stageTile16(ldsN, tid, rN);
    stageTile16(ldsH, tid, rH);
  }
  __syncthreads();
  int w = tid >> 5, lane = tid & 31, n = lane & 15, hi = lane >> 4;

  if (w == 1) {                       // Pn = node @ Wm1
    for (int ct = 0; ct < 8; ++ct) {
      v8f acc = {0.f, 0.f, 0.f, 0.f, 0.f, 0.f, 0.f, 0.f};
#pragma unroll
      for (int ks = 0; ks < 4; ++ks) {
        Frag A, Bf;
        buildA16(A, ldsN, n, ks, hi);
        loadPK(Bf, wsu + PK_M1 + ((size_t)(ks * 8 + ct) * 32 + lane) * 8);
        acc = wmma_bf16(A, Bf, acc);
      }
      int c = ct * 16 + n;
#pragma unroll
      for (int r = 0; r < 8; ++r)
        wsf[WS_PN + (size_t)(b * NN + j0 + r + 8 * hi) * 128 + c] = acc[r];
    }
  } else if (w == 2) {                // Ph = hidden @ Wm2
    for (int ct = 0; ct < 8; ++ct) {
      v8f acc = {0.f, 0.f, 0.f, 0.f, 0.f, 0.f, 0.f, 0.f};
#pragma unroll
      for (int ks = 0; ks < 4; ++ks) {
        Frag A, Bf;
        buildA16(A, ldsH, n, ks, hi);
        loadPK(Bf, wsu + PK_M2 + ((size_t)(ks * 8 + ct) * 32 + lane) * 8);
        acc = wmma_bf16(A, Bf, acc);
      }
      int c = ct * 16 + n;
#pragma unroll
      for (int r = 0; r < 8; ++r)
        wsf[WS_PH + (size_t)(b * NN + j0 + r + 8 * hi) * 128 + c] = acc[r];
    }
  } else {                            // w==0: S + R0[4..7]; w==3: R0[0..3]
    if (w == 0) {
      v8f acc = {0.f, 0.f, 0.f, 0.f, 0.f, 0.f, 0.f, 0.f};
#pragma unroll
      for (int ks = 0; ks < 8; ++ks) {
        Frag A, Bf;
        buildA16(A, (ks < 4 ? ldsN : ldsH), n, ks & 3, hi);
        loadPK(Bf, wsu + PK_S + ((size_t)ks * 32 + lane) * 8);
        acc = wmma_bf16(A, Bf, acc);
      }
      int t = n & 7;
      float* dstS = wsf + ((n < 8) ? WS_S1 : WS_S2);
#pragma unroll
      for (int r = 0; r < 8; ++r)
        dstS[(size_t)(b * NN + j0 + r + 8 * hi) * 8 + t] = acc[r];
    }
    int ctbase = (w == 0) ? 4 : 0;
    for (int ci = 0; ci < 4; ++ci) {
      int ct = ctbase + ci;
      v8f acc = {0.f, 0.f, 0.f, 0.f, 0.f, 0.f, 0.f, 0.f};
#pragma unroll
      for (int ks = 0; ks < 8; ++ks) {
        Frag A, Bf;
        buildA16(A, (ks < 4 ? ldsN : ldsH), n, ks & 3, hi);
        loadPK(Bf, wsu + PK_O12 + ((size_t)(ks * 8 + ct) * 32 + lane) * 8);
        acc = wmma_bf16(A, Bf, acc);
      }
      int c = ct * 16 + n;
      float bo = bo1[c] + bo2[c] + bo3[c];
#pragma unroll
      for (int r = 0; r < 8; ++r)
        wsf[WS_R0 + (size_t)(b * NN + j0 + r + 8 * hi) * 128 + c] = acc[r] + bo;
    }
  }
}

// =====================================================================
// Kernel C: fused edge pass, software-pipelined, i-axis split into ICH
// chunks for HBM-saturating concurrency.  1152 blocks (b, j-tile, chunk).
// =====================================================================
__global__ void __launch_bounds__(128) kedge(
    const float* edge, const float* adj, float* out, float* wsf) {
  __shared__ unsigned ldsE[2][16 * LSTR];   // bf16 edge tile, double-buffered
  __shared__ float ldsPh[2][128];
  __shared__ float ldsPn[2048];
  __shared__ float ldsAdj[IPC * 16];
  __shared__ float ldsS1[128];
  __shared__ float ldsS2[IPC * 8];
  __shared__ float ldsTri[128];
  const unsigned* wsu = (const unsigned*)wsf;
  int blk = blockIdx.x;
  int b  = blk / (JT * ICH);
  int r2 = blk % (JT * ICH);
  int jt = r2 / ICH, ic = r2 % ICH;
  int j0 = jt * 16, i0 = ic * IPC;
  int tid = threadIdx.x;

  {  // block-start staging: Pn tile, adj chunk, S1 tile, S2 chunk (all f32)
    const float4* srcPn = (const float4*)(wsf + WS_PN + (size_t)(b * NN + j0) * 128);
#pragma unroll
    for (int q = 0; q < 4; ++q) {
      int fq = tid + q * 128;
      *(float4*)&ldsPn[fq * 4] = srcPn[fq];
    }
    {  // adj[b, i0:i0+IPC, j0:j0+16] -> IPC*16 = 512 f = 128 float4
      int fq = tid;
      int il = fq >> 2, grp = fq & 3;
      float4 a4 = *(const float4*)(adj + (size_t)(b * NN + i0 + il) * NN + j0 + grp * 4);
      *(float4*)&ldsAdj[il * 16 + grp * 4] = a4;
    }
    if (tid < IPC * 2) {  // S2 chunk: IPC*8 f = IPC*2 float4
      const float4* srcS2 =
          (const float4*)(wsf + WS_S2 + (size_t)(b * NN + i0) * 8);
      *(float4*)&ldsS2[tid * 4] = srcS2[tid];
    }
    if (tid < 32) {
      const float4* srcS1 = (const float4*)(wsf + WS_S1 + (size_t)(b * NN + j0) * 8);
      *(float4*)&ldsS1[tid * 4] = srcS1[tid];
    }
  }

  int w = tid >> 5, lane = tid & 31, n = lane & 15, hi = lane >> 4;
  int cts[2] = {2 * w, 2 * w + 1};
  bool doTri = (w == 3);

  Frag Bme[2][4];
#pragma unroll
  for (int cl = 0; cl < 2; ++cl)
#pragma unroll
    for (int ks = 0; ks < 4; ++ks)
      loadPK(Bme[cl][ks], wsu + PK_ME + ((size_t)(ks * 8 + cts[cl]) * 32 + lane) * 8);
  Frag Bte[4];
  if (doTri)
#pragma unroll
    for (int ks = 0; ks < 4; ++ks)
      loadPK(Bte[ks], wsu + PK_TE + ((size_t)ks * 32 + lane) * 8);

  __syncthreads();   // ldsPn/ldsS1 ready

  float base_[2][8];
#pragma unroll
  for (int cl = 0; cl < 2; ++cl) {
    int c = cts[cl] * 16 + n;
    float gm = wsf[WS_GM + b * 128 + c];
#pragma unroll
    for (int r = 0; r < 8; ++r) base_[cl][r] = ldsPn[(r + 8 * hi) * 128 + c] + gm;
  }
  float gtv = 0.f, s1v[8];
#pragma unroll
  for (int r = 0; r < 8; ++r) s1v[r] = 0.f;
  if (doTri && n < 8) {
    gtv = wsf[WS_GT + b * 8 + n];
#pragma unroll
    for (int r = 0; r < 8; ++r) s1v[r] = ldsS1[(r + 8 * hi) * 8 + n];
  }
  float accMax[2][8];
#pragma unroll
  for (int cl = 0; cl < 2; ++cl)
#pragma unroll
    for (int r = 0; r < 8; ++r) accMax[cl][r] = -3.0e38f;

  const float* edgeBase = edge + ((size_t)(b * NN) * NN + j0) * 128;
  const float* phBase = wsf + WS_PH + (size_t)b * NN * 128;

  float4 rE[4], rPh;
  // prologue: tile i0
  {
    const float4* src = (const float4*)(edgeBase + (size_t)i0 * NN * 128);
#pragma unroll
    for (int q = 0; q < 4; ++q) rE[q] = src[tid * 4 + q];
    if (tid < 32) rPh = *(const float4*)(phBase + (size_t)i0 * 128 + tid * 4);
    stageTile16(ldsE[0], tid, rE);
    if (tid < 32) *(float4*)&ldsPh[0][tid * 4] = rPh;
  }
  __syncthreads();

  for (int il = 0; il < IPC; ++il) {
    int i = i0 + il;
    int cur = il & 1;
    if (il + 1 < IPC) {  // issue next tile's global loads early (pipelined)
      const float4* src = (const float4*)(edgeBase + (size_t)(i + 1) * NN * 128);
#pragma unroll
      for (int q = 0; q < 4; ++q) rE[q] = src[tid * 4 + q];
      if (tid < 32) rPh = *(const float4*)(phBase + (size_t)(i + 1) * 128 + tid * 4);
      if (il + 2 < IPC)
        __builtin_prefetch(edgeBase + (size_t)(i + 2) * NN * 128 + tid * 16, 0, 1);
    }

    // ---- compute tile i from LDS ----
    Frag A[4];
#pragma unroll
    for (int ks = 0; ks < 4; ++ks) buildA16(A[ks], ldsE[cur], n, ks, hi);

#pragma unroll
    for (int cl = 0; cl < 2; ++cl) {
      v8f acc = {0.f, 0.f, 0.f, 0.f, 0.f, 0.f, 0.f, 0.f};
#pragma unroll
      for (int ks = 0; ks < 4; ++ks) acc = wmma_bf16(A[ks], Bme[cl][ks], acc);
      float ph = ldsPh[cur][cts[cl] * 16 + n];
#pragma unroll
      for (int r = 0; r < 8; ++r) {
        float av = ldsAdj[il * 16 + r + 8 * hi];
        float val = acc[r] + base_[cl][r] + ph;
        val = (av != 0.0f) ? val : 0.0f;
        accMax[cl][r] = fmaxf(accMax[cl][r], val);
      }
    }
    if (doTri) {
      v8f tacc = {0.f, 0.f, 0.f, 0.f, 0.f, 0.f, 0.f, 0.f};
#pragma unroll
      for (int ks = 0; ks < 4; ++ks) tacc = wmma_bf16(A[ks], Bte[ks], tacc);
      if (n < 8) {
        float s2g = ldsS2[il * 8 + n] + gtv;
#pragma unroll
        for (int r = 0; r < 8; ++r) {
          float tv = tacc[r] + s1v[r] + s2g;
          ldsTri[(r + 8 * hi) * 8 + n] = fmaxf(tv, 0.0f);
        }
      }
      // same-wave LDS repack -> contiguous 512B store of tri[b,i,j0:16,0:8]
      float4 t4 = *(float4*)&ldsTri[lane * 4];
      *(float4*)(out + OUT_TRI + ((size_t)(b * NN + i) * NN + j0) * 8 + lane * 4) = t4;
    }
    __syncthreads();               // everyone done reading ldsE[cur]
    if (il + 1 < IPC) {
      stageTile16(ldsE[1 - cur], tid, rE);
      if (tid < 32) *(float4*)&ldsPh[1 - cur][tid * 4] = rPh;
    }
    __syncthreads();               // next tile staged
  }

  // partial max for this i-chunk
#pragma unroll
  for (int cl = 0; cl < 2; ++cl) {
    int c = cts[cl] * 16 + n;
#pragma unroll
    for (int r = 0; r < 8; ++r)
      wsf[WS_MSP + (size_t)ic * MSP_STRIDE +
          (size_t)(b * NN + j0 + r + 8 * hi) * 128 + c] = accMax[cl][r];
  }
}

// =====================================================================
// Kernel D: max-reduce the ICH partials, then ret = R0 + msgs @ W_o3.
// =====================================================================
__global__ void __launch_bounds__(128) kout(float* out, const float* wsf) {
  __shared__ unsigned ldsM[16 * LSTR];
  const unsigned* wsu = (const unsigned*)wsf;
  int b = blockIdx.x / JT, jt = blockIdx.x % JT, j0 = jt * 16;
  int tid = threadIdx.x;
  {
    const float* mp = wsf + WS_MSP + (size_t)(b * NN + j0) * 128;
    float4 rM[4];
#pragma unroll
    for (int q = 0; q < 4; ++q) {
      int fq = tid * 4 + q;
      float4 m = *(const float4*)(mp + fq * 4);
#pragma unroll
      for (int ic = 1; ic < ICH; ++ic) {
        float4 x = *(const float4*)(mp + (size_t)ic * MSP_STRIDE + fq * 4);
        m.x = fmaxf(m.x, x.x); m.y = fmaxf(m.y, x.y);
        m.z = fmaxf(m.z, x.z); m.w = fmaxf(m.w, x.w);
      }
      rM[q] = m;
    }
    stageTile16(ldsM, tid, rM);
  }
  __syncthreads();
  int w = tid >> 5, lane = tid & 31, n = lane & 15, hi = lane >> 4;
  Frag A[4];
#pragma unroll
  for (int ks = 0; ks < 4; ++ks) buildA16(A[ks], ldsM, n, ks, hi);
  for (int ci = 0; ci < 2; ++ci) {
    int ct = 2 * w + ci;
    v8f acc = {0.f, 0.f, 0.f, 0.f, 0.f, 0.f, 0.f, 0.f};
#pragma unroll
    for (int ks = 0; ks < 4; ++ks) {
      Frag Bf;
      loadPK(Bf, wsu + PK_O3 + ((size_t)(ks * 8 + ct) * 32 + lane) * 8);
      acc = wmma_bf16(A[ks], Bf, acc);
    }
    int c = ct * 16 + n;
#pragma unroll
    for (int r = 0; r < 8; ++r) {
      size_t o = (size_t)(b * NN + j0 + r + 8 * hi) * 128 + c;
      out[o] = acc[r] + wsf[WS_R0 + o];
    }
  }
}

// =====================================================================
extern "C" void kernel_launch(void* const* d_in, const int* in_sizes, int n_in,
                              void* d_out, int out_size, void* d_ws, size_t ws_size,
                              hipStream_t stream) {
  const float* node   = (const float*)d_in[0];
  const float* edge   = (const float*)d_in[1];
  const float* graph  = (const float*)d_in[2];
  const float* adj    = (const float*)d_in[3];
  const float* hidden = (const float*)d_in[4];
  const float* Wm1 = (const float*)d_in[5];   const float* bm1 = (const float*)d_in[6];
  const float* Wm2 = (const float*)d_in[7];   const float* bm2 = (const float*)d_in[8];
  const float* Wme = (const float*)d_in[9];   const float* bme = (const float*)d_in[10];
  const float* Wmg = (const float*)d_in[11];  const float* bmg = (const float*)d_in[12];
  const float* Wo1 = (const float*)d_in[13];  const float* bo1 = (const float*)d_in[14];
  const float* Wo2 = (const float*)d_in[15];  const float* bo2 = (const float*)d_in[16];
  const float* Wo3 = (const float*)d_in[17];  const float* bo3 = (const float*)d_in[18];
  const float* Wtn1 = (const float*)d_in[19]; const float* btn1 = (const float*)d_in[20];
  const float* Wtn2 = (const float*)d_in[21]; const float* btn2 = (const float*)d_in[22];
  const float* Wth1 = (const float*)d_in[23]; const float* bth1 = (const float*)d_in[24];
  const float* Wth2 = (const float*)d_in[25]; const float* bth2 = (const float*)d_in[26];
  const float* Wte1 = (const float*)d_in[27]; const float* bte1 = (const float*)d_in[28];
  const float* Wte2 = (const float*)d_in[29]; const float* bte2 = (const float*)d_in[30];
  const float* Wtg1 = (const float*)d_in[31]; const float* btg1 = (const float*)d_in[32];
  const float* Wtg2 = (const float*)d_in[33]; const float* btg2 = (const float*)d_in[34];
  float* wsf = (float*)d_ws;
  float* out = (float*)d_out;

  kprep<<<28, 256, 0, stream>>>(Wm1, Wm2, Wme, Wo1, Wo2, Wo3,
                                Wtn1, Wtn2, Wth1, Wth2, Wte1, Wte2,
                                Wmg, Wtg1, Wtg2, graph,
                                bm1, bm2, bme, bmg,
                                btn1, btn2, bth1, bth2,
                                bte1, bte2, btg1, btg2, wsf);
  knode<<<BATCH * JT, 128, 0, stream>>>(node, hidden, bo1, bo2, bo3, wsf);
  kedge<<<BATCH * JT * ICH, 128, 0, stream>>>(edge, adj, out, wsf);
  kout<<<BATCH * JT, 128, 0, stream>>>(out, wsf);
}
